// W_MSA_29618094473918
// MI455X (gfx1250) — matrix operand order repaired and checked by reading, not compiled, not run on hardware
//
#include <hip/hip_runtime.h>

// ---------------------------------------------------------------------------
// W-MSA (Swin window attention) fused kernel for gfx1250 (MI455X, wave32).
// One workgroup per 8x8 window; all intermediates live in LDS; all GEMMs use
// v_wmma_f32_16x16x32_f16.
// ---------------------------------------------------------------------------

typedef __attribute__((ext_vector_type(16))) _Float16 v16h;
typedef __attribute__((ext_vector_type(8)))  _Float16 v8h;
typedef __attribute__((ext_vector_type(8)))  float    v8f;

#define DIM      384
#define NH       12
#define HD       32
#define WN       64          // tokens per 8x8 window
#define NTHREADS 256
#define NWAVES   8

// ---- LDS carve (bytes). Total 285,696 < 320KB/WGP. ----
#define STR_X    392                         // halfs; %8==0 -> 16B aligned rows
#define OFF_X    0                           // 64 x 392 f16 : X (phase1) / O (phase2+)
#define STR_QK   48                          // halfs; %16==0 -> 32B aligned rows
#define OFF_Q    50176                       // 12 x 64 x 48 f16
#define OFF_K    123904                      // 12 x 64 x 48 f16
#define STR_VT   80                          // halfs; %16==0
#define OFF_V    197632                      // 12 x 32 x 80 f16  (V^T per head)
#define STR_S    68                          // f32
#define OFF_S    259072                      // 64 x 68 f32 scores
#define STR_P    72                          // halfs; %8==0
#define OFF_P    276480                      // 64 x 72 f16 probs
#define SMEM_BYTES 285696
#define OFF_F    OFF_Q                       // aliased: 384 x 68 f32 proj staging
#define STR_F    68

// ---------------------------------------------------------------------------
// Fragment loaders (contiguous k order, same for A and B; wave32)
// ---------------------------------------------------------------------------

// A fragment (16x32 f16): lane l holds row (l&15); K = kb + ((l&16)?16:0) + j,
// j = 0..15. Two contiguous 16B LDS loads.
__device__ inline v16h frag_a_lds(const _Float16* base, int stride, int kb) {
  int l = threadIdx.x & 31;
  const _Float16* p = base + (l & 15) * stride + kb + ((l & 16) ? 16 : 0);
  v8h lo = *(const v8h*)p;
  v8h hi = *(const v8h*)(p + 8);
  return __builtin_shufflevector(lo, hi, 0,1,2,3,4,5,6,7,8,9,10,11,12,13,14,15);
}

// B fragment (32x16 f16): lane l holds column (l&15); K = kb + ((l&16)?16:0) + j.
// Source stored row-major [col][k] (i.e. B^T), so 16 contiguous halfs per lane.
__device__ inline v16h frag_b_lds(const _Float16* base, int stride, int kb) {
  int l = threadIdx.x & 31;
  const _Float16* p = base + (l & 15) * stride + kb + ((l & 16) ? 16 : 0);
  v8h lo = *(const v8h*)p;
  v8h hi = *(const v8h*)(p + 8);
  return __builtin_shufflevector(lo, hi, 0,1,2,3,4,5,6,7,8,9,10,11,12,13,14,15);
}

__device__ inline v16h frag_b_glb(const _Float16* base, int stride, int kb) {
  int l = threadIdx.x & 31;
  return *(const v16h*)(base + (size_t)(l & 15) * stride + kb + ((l & 16) ? 16 : 0));
}

__device__ inline v8f wmma_f16(v16h a, v16h b, v8f c) {
  return __builtin_amdgcn_wmma_f32_16x16x32_f16(false, a, false, b, (short)0, c,
                                                false, false);
}

// GEMM inner loop: A from LDS (hoistable), B streamed from global + prefetch.
__device__ inline v8f gemm_row(const _Float16* a0, int astr,
                               const _Float16* b0) {
  v8f acc = {};
  #pragma unroll
  for (int kb = 0; kb < DIM; kb += 32) {
    __builtin_prefetch(b0 + 2 * 16 * DIM + kb * 16, 0, 1);  // next o-tile panel
    acc = wmma_f16(frag_a_lds(a0, astr, kb), frag_b_glb(b0, DIM, kb), acc);
  }
  return acc;
}

// ---------------------------------------------------------------------------
// Prep kernels: weight f32->f16, bias gather (12,64,64)
// ---------------------------------------------------------------------------

__global__ void wmsa_cvt_f16_kernel(const float* __restrict__ in,
                                    _Float16* __restrict__ out, int n) {
  int i = blockIdx.x * blockDim.x + threadIdx.x;
  if (i < n) out[i] = (_Float16)in[i];
}

__global__ void wmsa_build_bias_kernel(const float* __restrict__ rpb,
                                       const int* __restrict__ rel,
                                       float* __restrict__ biasbuf) {
  int i = blockIdx.x * blockDim.x + threadIdx.x;   // n*64+m, 0..4095
  if (i < WN * WN) {
    int r = rel[i];
    #pragma unroll
    for (int h = 0; h < NH; ++h)
      biasbuf[h * WN * WN + i] = rpb[r * NH + h];
  }
}

// ---------------------------------------------------------------------------
// Fused W-MSA kernel: one block = one window, 256 threads = 8 wave32.
// ---------------------------------------------------------------------------

__global__ void __launch_bounds__(NTHREADS)
wmsa_main_kernel(const float* __restrict__ x,          // (16,384,64,64)
                 const _Float16* __restrict__ Wq,      // (1152,384) f16
                 const float* __restrict__ qkv_b,      // (1152,)
                 const _Float16* __restrict__ Wp,      // (384,384) f16
                 const float* __restrict__ proj_b,     // (384,)
                 const float* __restrict__ biasbuf,    // (12,64,64) f32
                 float* __restrict__ out)              // (16,384,64,64)
{
  extern __shared__ char smem[];
  _Float16* Xs = (_Float16*)(smem + OFF_X);
  _Float16* Qs = (_Float16*)(smem + OFF_Q);
  _Float16* Ks = (_Float16*)(smem + OFF_K);
  _Float16* Vt = (_Float16*)(smem + OFF_V);
  float*    Ss = (float*)(smem + OFF_S);
  _Float16* Ps = (_Float16*)(smem + OFF_P);
  float*    Fs = (float*)(smem + OFF_F);

  const int tid  = threadIdx.x;
  const int wave = tid >> 5;
  const int lane = tid & 31;
  const int hi8  = (lane & 16) ? 8 : 0;     // C/D row split within lane groups
  const int col  = lane & 15;

  const int bw   = blockIdx.x;              // window id, 0..1023
  const int bimg = bw >> 6;
  const int wid  = bw & 63;
  const int h0   = (wid >> 3) << 3;
  const int w0   = (wid & 7) << 3;
  const float* xb = x + (size_t)bimg * DIM * 64 * 64;

  // Per-wave fixed tile coordinates: over t = wave + 8k (288 tiles, 8 waves),
  // tm = t&3 = wave&3 is constant and to = t>>2 = (wave>>2) + 2k.
  const int tm = wave & 3;                  // token tile (fixed per wave)
  const int pa = wave >> 2;                 // o-tile parity offset (0 or 1)
  const int dlo = ((pa & 1) << 4) + col;    // d within head for even/odd o-tiles

  // -------- Phase 0: window partition, f32 -> f16, LDS [token][c] ----------
  for (int idx = tid; idx < WN * DIM; idx += NTHREADS) {
    int token = idx & 63;
    int c     = idx >> 6;
    float v = xb[((size_t)c << 12) + ((h0 + (token >> 3)) << 6) + w0 + (token & 7)];
    Xs[token * STR_X + c] = (_Float16)v;
  }
  __syncthreads();

  // -------- Phase 1: QKV GEMM (64x384)x(384x1152), split to Q/K/V^T --------
  const float scale = 0.17677669529663687f;  // hd^-0.5
  const _Float16* a0 = Xs + tm * 16 * STR_X;

  // Q tiles: to in [0,24)
  for (int to = pa; to < 24; to += 2) {
    v8f acc = gemm_row(a0, STR_X, Wq + (size_t)to * 16 * DIM);
    float bv = qkv_b[to * 16 + col];
    int hh = to >> 1;
    _Float16* dst = Qs + hh * (WN * STR_QK) + dlo;
    #pragma unroll
    for (int r = 0; r < 8; ++r)
      dst[(tm * 16 + r + hi8) * STR_QK] = (_Float16)((acc[r] + bv) * scale);
  }
  // K tiles: to in [24,48)
  for (int to = 24 + pa; to < 48; to += 2) {
    v8f acc = gemm_row(a0, STR_X, Wq + (size_t)to * 16 * DIM);
    float bv = qkv_b[to * 16 + col];
    int hh = (to - 24) >> 1;
    _Float16* dst = Ks + hh * (WN * STR_QK) + dlo;
    #pragma unroll
    for (int r = 0; r < 8; ++r)
      dst[(tm * 16 + r + hi8) * STR_QK] = (_Float16)(acc[r] + bv);
  }
  // V tiles: to in [48,72), stored transposed [d][token]
  for (int to = 48 + pa; to < 72; to += 2) {
    v8f acc = gemm_row(a0, STR_X, Wq + (size_t)to * 16 * DIM);
    float bv = qkv_b[to * 16 + col];
    int hh = (to - 48) >> 1;
    _Float16* dst = Vt + hh * (HD * STR_VT) + dlo * STR_VT + tm * 16 + hi8;
    #pragma unroll
    for (int r = 0; r < 8; ++r)
      dst[r] = (_Float16)(acc[r] + bv);
  }
  __syncthreads();

  // -------- Phase 2: per-head attention ------------------------------------
  for (int h = 0; h < NH; ++h) {
    const _Float16* Qh = Qs + h * WN * STR_QK;
    const _Float16* Kh = Ks + h * WN * STR_QK;
    const _Float16* Vh = Vt + h * HD * STR_VT;

    // S = (Q*scale) K^T : 16 tiles; per wave tn fixed -> hoist K fragment.
    {
      const int tn = wave & 3;
      v16h bfrag = frag_b_lds(Kh + tn * 16 * STR_QK, STR_QK, 0);
      #pragma unroll
      for (int ti = 0; ti < 2; ++ti) {
        int tq = (wave >> 2) + 2 * ti;      // query tile 0..3
        v8f acc = {};
        acc = wmma_f16(frag_a_lds(Qh + tq * 16 * STR_QK, STR_QK, 0), bfrag, acc);
        #pragma unroll
        for (int r = 0; r < 8; ++r)
          Ss[(tq * 16 + r + hi8) * STR_S + tn * 16 + col] = acc[r];
      }
    }
    __syncthreads();

    // softmax rows (one thread per query row), + relative position bias
    if (tid < WN) {
      const float* bh = biasbuf + h * WN * WN + tid * WN;
      float mx = -1e30f;
      for (int j = 0; j < WN; ++j)
        mx = fmaxf(mx, Ss[tid * STR_S + j] + bh[j]);
      float sum = 0.f;
      for (int j = 0; j < WN; ++j)
        sum += __expf(Ss[tid * STR_S + j] + bh[j] - mx);
      float inv = 1.0f / sum;
      for (int j = 0; j < WN; ++j)
        Ps[tid * STR_P + j] = (_Float16)(__expf(Ss[tid * STR_S + j] + bh[j] - mx) * inv);
    }
    __syncthreads();

    // O_h = P V : 8 tiles (4 token-tiles x 2 d-tiles), one per wave, K=64
    {
      int tq = wave >> 1, td = wave & 1;
      v8f acc = {};
      #pragma unroll
      for (int kb = 0; kb < WN; kb += 32)
        acc = wmma_f16(frag_a_lds(Ps + tq * 16 * STR_P, STR_P, kb),
                       frag_b_lds(Vh + td * 16 * STR_VT, STR_VT, kb), acc);
      #pragma unroll
      for (int r = 0; r < 8; ++r)
        Xs[(tq * 16 + r + hi8) * STR_X + h * HD + td * 16 + col] = (_Float16)acc[r];
    }
    __syncthreads();
  }

  // -------- Phase 3: output projection (64x384)x(384x384) ------------------
  for (int to = pa; to < 24; to += 2) {
    v8f acc = gemm_row(a0, STR_X, Wp + (size_t)to * 16 * DIM);  // O lives in Xs
    int c = to * 16 + col;
    float pb = proj_b[c];
    float* dst = Fs + c * STR_F + tm * 16 + hi8;
    #pragma unroll
    for (int r = 0; r < 8; ++r)
      dst[r] = acc[r] + pb;                 // staged [c][token]
  }
  __syncthreads();

  // -------- window reverse: coalesced f32 store ----------------------------
  float* ob = out + (size_t)bimg * DIM * 64 * 64;
  for (int idx = tid; idx < WN * DIM; idx += NTHREADS) {
    int token = idx & 63;
    int c     = idx >> 6;
    ob[((size_t)c << 12) + ((h0 + (token >> 3)) << 6) + w0 + (token & 7)] =
        Fs[c * STR_F + token];
  }
}

// ---------------------------------------------------------------------------
// Host-side launcher
// ---------------------------------------------------------------------------

extern "C" void kernel_launch(void* const* d_in, const int* in_sizes, int n_in,
                              void* d_out, int out_size, void* d_ws, size_t ws_size,
                              hipStream_t stream) {
  (void)in_sizes; (void)n_in; (void)out_size; (void)ws_size;
  const float* x      = (const float*)d_in[0];
  const float* qkv_w  = (const float*)d_in[1];
  const float* qkv_b  = (const float*)d_in[2];
  const float* proj_w = (const float*)d_in[3];
  const float* proj_b = (const float*)d_in[4];
  const float* rpb    = (const float*)d_in[5];
  const int*   rel    = (const int*)d_in[6];
  float* out = (float*)d_out;

  // workspace carve: Wq16 (1152x384 f16), Wp16 (384x384 f16), bias (12x64x64 f32)
  char* ws = (char*)d_ws;
  _Float16* Wq16 = (_Float16*)ws;
  _Float16* Wp16 = (_Float16*)(ws + (size_t)1152 * 384 * 2);
  float* biasbuf = (float*)(ws + (size_t)1152 * 384 * 2 + (size_t)384 * 384 * 2);

  const int nq = 1152 * 384;
  const int np = 384 * 384;
  wmsa_cvt_f16_kernel<<<(nq + 255) / 256, 256, 0, stream>>>(qkv_w, Wq16, nq);
  wmsa_cvt_f16_kernel<<<(np + 255) / 256, 256, 0, stream>>>(proj_w, Wp16, np);
  wmsa_build_bias_kernel<<<(WN * WN + 255) / 256, 256, 0, stream>>>(rpb, rel, biasbuf);

  wmsa_main_kernel<<<1024, NTHREADS, SMEM_BYTES, stream>>>(
      x, Wq16, qkv_b, Wp16, proj_b, biasbuf, out);
}
